// GAT_3882650435796
// MI455X (gfx1250) — matrix operand order repaired and checked by reading, not compiled, not run on hardware
//
#include <hip/hip_runtime.h>
#include <hip/hip_bf16.h>

// GAT forward for MI455X (gfx1250, wave32, WMMA), single-adj-pass version.
//
// Roofline: adj (8192^2 int32 = 256MB) is the dominant unavoidable traffic;
// at 23.3 TB/s that's ~11us. GEMM1 (2.1 GFLOP) + GEMM2 (34.4 GFLOP) run on
// v_wmma_f32_16x16x32_bf16 and are small against that. The [H,N,N] score
// tensor is rank-1 (f1[i]+f2[j]) and never materialized.
//
// Softmax trick: LeakyReLU is monotone increasing, so
//     M_i = leaky(f1[h,i] + max_j f2[h,j])
// upper-bounds every unmasked score e_ij = leaky(f1+f2). Softmax is shift
// invariant and exp(e - M_i) in (0,1], so M_i is a safe stabilizer computed
// from a per-head scalar — no adjacency pass needed for the max. The row sum
// is accumulated from the very A-tile weights fed to WMMA in GEMM2. Result:
// adjacency is streamed exactly once. (A fully-masked row would divide by ~0;
// with p=0.5 Bernoulli adjacency that has probability ~2^-8192.)
//
// Pipeline:
//  1) cvt_x:       x f32 -> bf16
//  2) transpose_w: W[h][f][o] -> Wt[h][o][f] bf16 (contiguous WMMA B loads)
//  3) gat_gemm1:   h_t[h][o][n] = (x@W)^T via bf16 WMMA
//  4) f1f2:        f1/f2[h][n] = h.a1 / h.a2
//  5) f2max:       per-head max of f2 (32KB reduction)
//  6) gat_attn_gemm2: A-tiles = exp weights built on the fly, B = h_t tiles,
//     4 waves split j, rowsum rides along, LDS reduce, /rowsum, sigmoid,
//     concat-transpose store.

typedef __bf16 bf16_t;
typedef __attribute__((ext_vector_type(16))) __bf16 v16bf;
typedef __attribute__((ext_vector_type(8)))  __bf16 v8bf;
typedef __attribute__((ext_vector_type(8)))  float  v8f;

constexpr int N     = 8192;
constexpr int F_IN  = 512;
constexpr int F_OUT = 64;
constexpr int H     = 4;
#define GAT_ALPHA 0.2f

__device__ inline v8f wmma_bf16(v16bf a, v16bf b, v8f c) {
  // (neg_a, A, neg_b, B, c_mod, C, reuse_a, reuse_b)
  return __builtin_amdgcn_wmma_f32_16x16x32_bf16(false, a, false, b, (short)0, c,
                                                 false, false);
}

// LeakyReLU via max: for alpha<1, leaky(e) == max(e, alpha*e).
__device__ inline float leaky(float e) { return fmaxf(e, GAT_ALPHA * e); }

// Assemble a v16bf from two contiguous 8-element loads (lowers to 2x b128).
__device__ inline v16bf pack16(const bf16_t* lo, const bf16_t* hi) {
  v8bf a = *(const v8bf*)lo;
  v8bf b = *(const v8bf*)hi;
  v16bf r;
#pragma unroll
  for (int e = 0; e < 8; ++e) { r[e] = a[e]; r[e + 8] = b[e]; }
  return r;
}

// ---------------- prep ----------------
__global__ void cvt_x_kernel(const float* __restrict__ x, bf16_t* __restrict__ xbf,
                             int n) {
  int i = blockIdx.x * blockDim.x + threadIdx.x;
  if (i < n) xbf[i] = (bf16_t)x[i];
}

__global__ void transpose_w_kernel(const float* __restrict__ W,
                                   bf16_t* __restrict__ wt) {
  int i = blockIdx.x * blockDim.x + threadIdx.x;  // over H*F_IN*F_OUT
  if (i >= H * F_IN * F_OUT) return;
  int o = i % F_OUT;
  int f = (i / F_OUT) % F_IN;
  int h = i / (F_OUT * F_IN);
  wt[((size_t)h * F_OUT + o) * F_IN + f] = (bf16_t)W[i];
}

// ---------------- GEMM1: h_t = (x @ W)^T, bf16 WMMA ----------------
__global__ void gat_gemm1_kernel(const bf16_t* __restrict__ xbf,   // [N][F_IN]
                                 const bf16_t* __restrict__ wtbf,  // [H][F_OUT][F_IN]
                                 bf16_t* __restrict__ h_t) {       // [H][F_OUT][N]
  const int wave = (int)((blockIdx.x * blockDim.x + threadIdx.x) >> 5);
  const int lane = threadIdx.x & 31;
  const int tiles_n = F_OUT / 16;  // 4
  const int tiles_m = N / 16;      // 512
  int t = wave;                    // H*512*4 = 8192 waves total
  const int head = t / (tiles_m * tiles_n);
  t %= tiles_m * tiles_n;
  const int m0 = (t / tiles_n) * 16;
  const int n0 = (t % tiles_n) * 16;

  const int hh  = lane >> 4;   // half-wave select
  const int l16 = lane & 15;

  const bf16_t* xrow = xbf + (size_t)(m0 + l16) * F_IN;                    // A row M
  const bf16_t* wrow = wtbf + ((size_t)head * F_OUT + (n0 + l16)) * F_IN;  // B col N

  v8f c = {};
#pragma unroll 4
  for (int k0 = 0; k0 < F_IN; k0 += 32) {
    // A 16x32: half-wave K split {0-7,16-23} / {8-15,24-31}
    v16bf a = pack16(xrow + k0 + hh * 8, xrow + k0 + 16 + hh * 8);
    // B 32x16: half-wave K split {0-7,16-23} / {8-15,24-31}, matching A
    v16bf b = pack16(wrow + k0 + hh * 8, wrow + k0 + 16 + hh * 8);
    c = wmma_bf16(a, b, c);
  }
  // C layout: lanes0-15 M=v, lanes16-31 M=8+v; N=l16. Store transposed.
#pragma unroll
  for (int v = 0; v < 8; ++v) {
    const int m = m0 + v + hh * 8;
    const int o = n0 + l16;
    h_t[((size_t)head * F_OUT + o) * N + m] = (bf16_t)c[v];
  }
}

// ---------------- f1/f2 ----------------
__global__ void f1f2_kernel(const bf16_t* __restrict__ h_t,
                            const float* __restrict__ a1,
                            const float* __restrict__ a2,
                            float* __restrict__ f1, float* __restrict__ f2) {
  int i = blockIdx.x * blockDim.x + threadIdx.x;  // over H*N
  if (i >= H * N) return;
  int n = i % N;
  int h = i / N;
  float s1 = 0.f, s2 = 0.f;
#pragma unroll 8
  for (int o = 0; o < F_OUT; ++o) {
    float hv = (float)h_t[((size_t)h * F_OUT + o) * N + n];  // coalesced across lanes
    s1 += hv * a1[h * F_OUT + o];
    s2 += hv * a2[h * F_OUT + o];
  }
  f1[i] = s1;
  f2[i] = s2;
}

// ---------------- per-head max of f2 ----------------
__global__ void f2max_kernel(const float* __restrict__ f2,
                             float* __restrict__ f2max) {
  __shared__ float red[8];
  const int h = blockIdx.x;       // one block per head
  const int t = threadIdx.x;      // 256 threads = 8 waves
  float m = -3.0e38f;
  for (int j = t; j < N; j += 256) m = fmaxf(m, f2[h * N + j]);
#pragma unroll
  for (int off = 16; off; off >>= 1) m = fmaxf(m, __shfl_xor(m, off, 32));
  if ((t & 31) == 0) red[t >> 5] = m;
  __syncthreads();
  if (t == 0) {
    float r = red[0];
#pragma unroll
    for (int w = 1; w < 8; ++w) r = fmaxf(r, red[w]);
    f2max[h] = r;
  }
}

// ------- attention GEMM2 + fused rowsum + sigmoid (single adj pass) -------
__global__ void gat_attn_gemm2_kernel(const int* __restrict__ adj,
                                      const bf16_t* __restrict__ h_t,  // [H][F_OUT][N]
                                      const float* __restrict__ f1,
                                      const float* __restrict__ f2,
                                      const float* __restrict__ f2max,
                                      float* __restrict__ out) {  // [N][H*F_OUT]
  __shared__ float red[4][4][8][32];  // [wave][o-tile][C vgpr][lane]
  __shared__ float sred[4][16];       // [wave][row-in-tile] partial rowsums
  const int lane = threadIdx.x & 31;
  const int wid  = threadIdx.x >> 5;  // 4 waves split the j range
  const int head = blockIdx.x / (N / 16);
  const int i0   = (blockIdx.x % (N / 16)) * 16;
  const int hh   = lane >> 4;
  const int l16  = lane & 15;

  // A rows: attention row i for this lane (A layout: M = lane&15)
  const int irow   = i0 + l16;
  const float f1v  = f1[head * N + irow];
  const float mv   = leaky(f1v + f2max[head]);  // monotone upper bound on row max
  const int* arow  = adj + (size_t)irow * N;
  const float* f2h = f2 + head * N;
  const bf16_t* hb = h_t + (size_t)head * F_OUT * N;

  v8f c0 = {}, c1 = {}, c2 = {}, c3 = {};
  float sacc = 0.f;  // this lane's partial sum of its 16 K-slots per tile
  for (int jt = wid; jt < N / 32; jt += 4) {
    const int j0 = jt * 32;
    if (j0 + 512 < N) __builtin_prefetch(arow + j0 + 512, 0, 0);  // global_prefetch_b8
    // Build A tile: unnormalized softmax weights, bf16.
    v16bf a;
#pragma unroll
    for (int e = 0; e < 8; ++e) {
      const int jlo = j0 + hh * 16 + e;       // K in {0-7}|{16-23}
      const int jhi = j0 + hh * 16 + 8 + e;   // K in {8-15}|{24-31}
      const float wlo =
          (arow[jlo] > 0) ? __expf(leaky(f1v + f2h[jlo]) - mv) : 0.f;
      const float whi =
          (arow[jhi] > 0) ? __expf(leaky(f1v + f2h[jhi]) - mv) : 0.f;
      sacc += wlo + whi;
      a[e]     = (bf16_t)wlo;
      a[e + 8] = (bf16_t)whi;
    }
    // B tiles: h_t column-contiguous per lane; 4 o-tiles share one A.
    const bf16_t* b0 = hb + (size_t)(0  + l16) * N + j0 + hh * 16;
    const bf16_t* b1 = hb + (size_t)(16 + l16) * N + j0 + hh * 16;
    const bf16_t* b2 = hb + (size_t)(32 + l16) * N + j0 + hh * 16;
    const bf16_t* b3 = hb + (size_t)(48 + l16) * N + j0 + hh * 16;
    c0 = wmma_bf16(a, pack16(b0, b0 + 8), c0);
    c1 = wmma_bf16(a, pack16(b1, b1 + 8), c1);
    c2 = wmma_bf16(a, pack16(b2, b2 + 8), c2);
    c3 = wmma_bf16(a, pack16(b3, b3 + 8), c3);
  }

  // Row-sum: lanes l and l^16 hold the two K-halves of row i0+l16.
  sacc += __shfl_xor(sacc, 16, 32);
  if (lane < 16) sred[wid][l16] = sacc;

  // Cross-wave partial reduction through LDS.
#pragma unroll
  for (int v = 0; v < 8; ++v) {
    red[wid][0][v][lane] = c0[v];
    red[wid][1][v][lane] = c1[v];
    red[wid][2][v][lane] = c2[v];
    red[wid][3][v][lane] = c3[v];
  }
  __syncthreads();
  if (wid == 0) {
#pragma unroll
    for (int ot = 0; ot < 4; ++ot) {
#pragma unroll
      for (int v = 0; v < 8; ++v) {
        const int r = v + hh * 8;             // row index within tile (C layout)
        const float acc = red[0][ot][v][lane] + red[1][ot][v][lane] +
                          red[2][ot][v][lane] + red[3][ot][v][lane];
        const float rs = sred[0][r] + sred[1][r] + sred[2][r] + sred[3][r];
        const int i = i0 + r;
        const int o = ot * 16 + l16;
        const float w  = acc / rs;
        const float sg = 1.f / (1.f + __expf(-w));
        out[(size_t)i * (H * F_OUT) + head * F_OUT + o] = sg;
      }
    }
  }
}

extern "C" void kernel_launch(void* const* d_in, const int* in_sizes, int n_in,
                              void* d_out, int out_size, void* d_ws, size_t ws_size,
                              hipStream_t stream) {
  const float* x   = (const float*)d_in[0];  // [N, F_IN]
  const int*   adj = (const int*)d_in[1];    // [N, N]
  const float* W   = (const float*)d_in[2];  // [H, F_IN, F_OUT]
  const float* a1  = (const float*)d_in[3];  // [H, F_OUT]
  const float* a2  = (const float*)d_in[4];  // [H, F_OUT]
  float* out = (float*)d_out;                // [N, H*F_OUT]

  char* ws = (char*)d_ws;
  bf16_t* xbf = (bf16_t*)ws;  ws += (size_t)N * F_IN * sizeof(bf16_t);        // 8 MB
  bf16_t* wt  = (bf16_t*)ws;  ws += (size_t)H * F_OUT * F_IN * sizeof(bf16_t);
  bf16_t* h_t = (bf16_t*)ws;  ws += (size_t)H * F_OUT * N * sizeof(bf16_t);   // 4 MB
  float* f1   = (float*)ws;   ws += (size_t)H * N * sizeof(float);
  float* f2   = (float*)ws;   ws += (size_t)H * N * sizeof(float);
  float* f2mx = (float*)ws;   ws += 256;  // H floats, padded

  cvt_x_kernel<<<(N * F_IN + 255) / 256, 256, 0, stream>>>(x, xbf, N * F_IN);
  transpose_w_kernel<<<(H * F_IN * F_OUT + 255) / 256, 256, 0, stream>>>(W, wt);
  // 8192 tile-waves, 4 waves (128 threads) per block
  gat_gemm1_kernel<<<(H * (N / 16) * (F_OUT / 16)) / 4, 128, 0, stream>>>(xbf, wt, h_t);
  f1f2_kernel<<<(H * N + 127) / 128, 128, 0, stream>>>(h_t, a1, a2, f1, f2);
  f2max_kernel<<<H, 256, 0, stream>>>(f2, f2mx);
  // one 4-wave block per (head, 16-row tile); adj streamed exactly once
  gat_attn_gemm2_kernel<<<H * (N / 16), 128, 0, stream>>>(adj, h_t, f1, f2, f2mx, out);
}
